// ImprovedGCN_43782896615863
// MI455X (gfx1250) — matrix operand order repaired and checked by reading, not compiled, not run on hardware
//
#include <hip/hip_runtime.h>
#include <hip/hip_bf16.h>

typedef __attribute__((ext_vector_type(16))) __bf16 v16bf;
typedef __attribute__((ext_vector_type(8)))  float  v8f;

#define NN   20000
#define EE   320000
#define INF_ 4644
#define HF   512
#define GN_EPS 1e-5f

// GEMM tiling
#define BM 128
#define BN 128
#define BK 32
#define LDA (BK + 8)
#define LDB (BK + 8)

__device__ __forceinline__ __bf16 to_bf16(float v)  { return (__bf16)v; }
__device__ __forceinline__ __bf16 to_bf16(__bf16 v) { return v; }

// 8 contiguous elements -> 8 bf16 via one 16B LDS store
__device__ __forceinline__ void load8_cvt(const float* p, __bf16* dst) {
  const float4 a = *(const float4*)p;
  const float4 b = *(const float4*)(p + 4);
  union { __bf16 h[8]; uint4 u; } t;
  t.h[0] = (__bf16)a.x; t.h[1] = (__bf16)a.y; t.h[2] = (__bf16)a.z; t.h[3] = (__bf16)a.w;
  t.h[4] = (__bf16)b.x; t.h[5] = (__bf16)b.y; t.h[6] = (__bf16)b.z; t.h[7] = (__bf16)b.w;
  *(uint4*)dst = t.u;
}
__device__ __forceinline__ void load8_cvt(const __bf16* p, __bf16* dst) {
  *(uint4*)dst = *(const uint4*)p;
}

// ---- A tile staging: BM x BK, 8 elems/chunk, 2 chunks/thread ----
template<typename AT, bool GR, bool GK>
__device__ __forceinline__ void stageA(const AT* __restrict__ A, __bf16* __restrict__ As,
                                       int bm0, int k0, int M, int K, int tid) {
#pragma unroll
  for (int c = tid; c < (BM * BK) / 8; c += 256) {
    const int row = c >> 2;
    const int kc  = (c & 3) * 8;
    const int gr  = bm0 + row;
    const int gk  = k0 + kc;
    __bf16* dst = &As[row * LDA + kc];
    const bool rowOk = !GR || (gr < M);
    if (!GK) {
      if (rowOk) load8_cvt(A + (long)gr * K + gk, dst);
      else       *(uint4*)dst = make_uint4(0, 0, 0, 0);
    } else {
      if (rowOk && (gk + 7) < K) {
        load8_cvt(A + (long)gr * K + gk, dst);
      } else {
#pragma unroll
        for (int e = 0; e < 8; ++e) {
          const int kk = gk + e;
          dst[e] = (rowOk && kk < K) ? to_bf16(A[(long)gr * K + kk]) : (__bf16)0.0f;
        }
      }
    }
  }
}

// ---- B tile staging from pre-transposed, pre-padded BT[NcPad][Kp]: no guards ----
__device__ __forceinline__ void stageB(const __bf16* __restrict__ BT, __bf16* __restrict__ Bs,
                                       int bn0, int k0, int Kp, int tid) {
#pragma unroll
  for (int c = tid; c < (BN * BK) / 8; c += 256) {
    const int col = c >> 2;
    const int kc  = (c & 3) * 8;
    *(uint4*)&Bs[col * LDB + kc] = *(const uint4*)&BT[(long)(bn0 + col) * Kp + k0 + kc];
  }
}

// ---------------------------------------------------------------------------
// WMMA GEMM: C[M,Nc] = A[M,K] * B[K,Nc], B supplied transposed+padded bf16.
// Block 128x128, K-step 32, double-buffered LDS, 8 waves, wave tile 32x64.
// ---------------------------------------------------------------------------
template<typename AT>
__global__ __launch_bounds__(256) void gemm_wmma_bf16(
    const AT* __restrict__ A, const __bf16* __restrict__ BT,
    int M, int K, int Kp, int Nc,
    const float* __restrict__ bias, int do_relu,
    float* __restrict__ outF, __bf16* __restrict__ outB)
{
  __shared__ __bf16 As[2 * BM * LDA];
  __shared__ __bf16 Bs[2 * BN * LDB];

  const int tid  = threadIdx.x;
  const int bm0  = blockIdx.y * BM;
  const int bn0  = blockIdx.x * BN;
  const int wave = tid >> 5;
  const int lane = tid & 31;
  const int wm   = wave & 3;    // 0..3 -> 32 rows each
  const int wn   = wave >> 2;   // 0..1 -> 64 cols each
  const int l16  = lane & 15;
  const int lh   = lane >> 4;

  const bool rowsFull = (bm0 + BM) <= M;
  const int  nsteps   = Kp / BK;

  // loop-invariant LDS fragment offsets (ISA bf16 A(16x32)/B(32x16) layouts)
  int aoff[2], boff[4];
#pragma unroll
  for (int s = 0; s < 2; ++s) aoff[s] = (wm * 32 + s * 16 + l16) * LDA + lh * 8;
#pragma unroll
  for (int t = 0; t < 4; ++t) boff[t] = (wn * 64 + t * 16 + l16) * LDB + lh * 8;

  v8f acc[2][4];
#pragma unroll
  for (int s = 0; s < 2; ++s)
#pragma unroll
    for (int t = 0; t < 4; ++t)
#pragma unroll
      for (int e = 0; e < 8; ++e) acc[s][t][e] = 0.0f;

  auto doStageA = [&](int buf, int k0) {
    __bf16* dstA = As + buf * (BM * LDA);
    if (rowsFull) {
      if (k0 + BK <= K) stageA<AT, false, false>(A, dstA, bm0, k0, M, K, tid);
      else              stageA<AT, false, true >(A, dstA, bm0, k0, M, K, tid);
    } else {
      if (k0 + BK <= K) stageA<AT, true, false>(A, dstA, bm0, k0, M, K, tid);
      else              stageA<AT, true, true >(A, dstA, bm0, k0, M, K, tid);
    }
  };

  doStageA(0, 0);
  stageB(BT, Bs, bn0, 0, Kp, tid);

  for (int i = 0; i < nsteps; ++i) {
    __syncthreads();
    if (i + 1 < nsteps) {
      const int kn = (i + 1) * BK;
      doStageA((i + 1) & 1, kn);
      stageB(BT, Bs + ((i + 1) & 1) * (BN * LDB), bn0, kn, Kp, tid);
    }
    const __bf16* Ab = As + (i & 1) * (BM * LDA);
    const __bf16* Bb = Bs + (i & 1) * (BN * LDB);

    v16bf afr[2], bfr[4];
#pragma unroll
    for (int s = 0; s < 2; ++s) {
      const __bf16* ap = Ab + aoff[s];
#pragma unroll
      for (int e = 0; e < 8; ++e) { afr[s][e] = ap[e]; afr[s][8 + e] = ap[16 + e]; }
    }
#pragma unroll
    for (int t = 0; t < 4; ++t) {
      const __bf16* bp = Bb + boff[t];
#pragma unroll
      for (int e = 0; e < 8; ++e) { bfr[t][e] = bp[e]; bfr[t][8 + e] = bp[16 + e]; }
    }
#pragma unroll
    for (int s = 0; s < 2; ++s)
#pragma unroll
      for (int t = 0; t < 4; ++t)
        acc[s][t] = __builtin_amdgcn_wmma_f32_16x16x32_bf16(
            false, afr[s], false, bfr[t], (short)0, acc[s][t], false, false);
  }

  // ---- epilogue ----
#pragma unroll
  for (int s = 0; s < 2; ++s)
#pragma unroll
    for (int t = 0; t < 4; ++t) {
      const int gc = bn0 + wn * 64 + t * 16 + l16;
      if (gc >= Nc) continue;
      const float bv = bias ? bias[gc] : 0.0f;
#pragma unroll
      for (int r = 0; r < 8; ++r) {
        const int gr = bm0 + wm * 32 + s * 16 + lh * 8 + r;
        if (gr >= M) continue;
        float v = acc[s][t][r] + bv;
        if (do_relu) v = fmaxf(v, 0.0f);
        const long idx = (long)gr * Nc + gc;
        if (outF) outF[idx] = v;
        if (outB) outB[idx] = (__bf16)v;
      }
    }
}

// ---------------------------------------------------------------------------
// Weight convert + transpose + pad:  in[K x Nc] f32 -> out[NcPad x Kp] bf16
// ---------------------------------------------------------------------------
__global__ void k_f2bf_T(const float* __restrict__ in, __bf16* __restrict__ out,
                         int K, int Nc, int Kp, int NcPad) {
  long i = (long)blockIdx.x * blockDim.x + threadIdx.x;
  const long total  = (long)NcPad * Kp;
  const long stride = (long)gridDim.x * blockDim.x;
  for (; i < total; i += stride) {
    const int col = (int)(i / Kp);
    const int k   = (int)(i % Kp);
    out[i] = (col < Nc && k < K) ? (__bf16)in[(long)k * Nc + col] : (__bf16)0.0f;
  }
}

// ---------------------------------------------------------------------------
// Graph / elementwise kernels (fp32)
// ---------------------------------------------------------------------------
__global__ void k_degree(const int* __restrict__ dst, float* __restrict__ deg, int E) {
  int e = blockIdx.x * blockDim.x + threadIdx.x;
  if (e < E) atomicAdd(&deg[dst[e]], 1.0f);
}

__global__ void k_dinv(const float* __restrict__ deg, float* __restrict__ dinv,
                       float* __restrict__ dinv2, int n) {
  int i = blockIdx.x * blockDim.x + threadIdx.x;
  if (i < n) {
    float d = deg[i] + 1.0f;
    dinv[i]  = rsqrtf(d);
    dinv2[i] = 1.0f / d;
  }
}

__global__ void k_coef(const int* __restrict__ src, const int* __restrict__ dst,
                       const float* __restrict__ dinv, float* __restrict__ coef, int E) {
  int e = blockIdx.x * blockDim.x + threadIdx.x;
  if (e < E) coef[e] = dinv[src[e]] * dinv[dst[e]];
}

// one block per edge; h (=xW, 41MB) is L2-resident
__global__ __launch_bounds__(256) void k_scatter(
    const int* __restrict__ src, const int* __restrict__ dst,
    const float* __restrict__ coef, const float* __restrict__ h,
    float* __restrict__ agg, int E)
{
  int e = blockIdx.x;
  if (e >= E) return;
  const int   s = src[e], d = dst[e];
  const float c = coef[e];
  const float* hs = h + (long)s * HF;
  float*       ad = agg + (long)d * HF;
  for (int ch = threadIdx.x; ch < HF; ch += 256)
    atomicAdd(&ad[ch], hs[ch] * c);
}

// y = agg + hW*dinv2 + bias (in place) + per-column partial sums
__global__ __launch_bounds__(256) void k_combine_colsum(
    float* __restrict__ y, const float* __restrict__ hW,
    const float* __restrict__ dinv2, const float* __restrict__ bias,
    float* __restrict__ colsum, int M)
{
  const int r0 = blockIdx.x * 32;
  const int c0 = threadIdx.x * 2;
  const float b0 = bias[c0], b1 = bias[c0 + 1];
  float s0 = 0.f, s1 = 0.f;
  const int rend = (r0 + 32 < M) ? r0 + 32 : M;
  for (int r = r0; r < rend; ++r) {
    const long base = (long)r * HF;
    const float d2 = dinv2[r];
    const float v0 = y[base + c0]     + hW[base + c0]     * d2 + b0;
    const float v1 = y[base + c0 + 1] + hW[base + c0 + 1] * d2 + b1;
    y[base + c0] = v0; y[base + c0 + 1] = v1;
    s0 += v0; s1 += v1;
  }
  atomicAdd(&colsum[c0], s0);
  atomicAdd(&colsum[c0 + 1], s1);
}

__global__ void k_mean(const float* __restrict__ colsum, float* __restrict__ mean) {
  int c = blockIdx.x * blockDim.x + threadIdx.x;
  if (c < HF) mean[c] = colsum[c] * (1.0f / (float)NN);
}

__global__ __launch_bounds__(256) void k_varsum(
    const float* __restrict__ y, const float* __restrict__ mean,
    const float* __restrict__ a, float* __restrict__ varsum, int M)
{
  const int r0 = blockIdx.x * 32;
  const int c0 = threadIdx.x * 2;
  const float am0 = a[c0] * mean[c0], am1 = a[c0 + 1] * mean[c0 + 1];
  float s0 = 0.f, s1 = 0.f;
  const int rend = (r0 + 32 < M) ? r0 + 32 : M;
  for (int r = r0; r < rend; ++r) {
    const long base = (long)r * HF;
    const float u0 = y[base + c0] - am0;
    const float u1 = y[base + c0 + 1] - am1;
    s0 += u0 * u0; s1 += u1 * u1;
  }
  atomicAdd(&varsum[c0], s0);
  atomicAdd(&varsum[c0 + 1], s1);
}

__global__ void k_rstd(const float* __restrict__ varsum, float* __restrict__ rstd) {
  int c = blockIdx.x * blockDim.x + threadIdx.x;
  if (c < HF) rstd[c] = rsqrtf(varsum[c] * (1.0f / (float)NN) + GN_EPS);
}

__global__ void k_norm_final(
    const float* __restrict__ y, const float* __restrict__ mean,
    const float* __restrict__ rstd, const float* __restrict__ a,
    const float* __restrict__ w, const float* __restrict__ b,
    const float* __restrict__ res, float* __restrict__ layer_out,
    __bf16* __restrict__ hnext_bf, long total)
{
  long i = (long)blockIdx.x * blockDim.x + threadIdx.x;
  long stride = (long)gridDim.x * blockDim.x;
  for (; i < total; i += stride) {
    const int c = (int)(i & (HF - 1));
    const float sub = y[i] - a[c] * mean[c];
    float ln = w[c] * sub * rstd[c] + b[c];
    ln = fmaxf(ln, 0.0f);
    layer_out[i] = ln;
    if (hnext_bf) {
      float hn = ln + (res ? res[i] : 0.0f);
      hnext_bf[i] = (__bf16)hn;
    }
  }
}

__global__ void k_jkmax_bf(const float* __restrict__ l0, const float* __restrict__ l1,
                           const float* __restrict__ l2, __bf16* __restrict__ out, long n) {
  long i = (long)blockIdx.x * blockDim.x + threadIdx.x;
  long stride = (long)gridDim.x * blockDim.x;
  for (; i < n; i += stride)
    out[i] = (__bf16)fmaxf(fmaxf(l0[i], l1[i]), l2[i]);
}

// ---------------------------------------------------------------------------
extern "C" void kernel_launch(void* const* d_in, const int* in_sizes, int n_in,
                              void* d_out, int out_size, void* d_ws, size_t ws_size,
                              hipStream_t stream) {
  (void)in_sizes; (void)n_in; (void)out_size; (void)ws_size;
  const float* x        = (const float*)d_in[0];
  const int*   ei       = (const int*)  d_in[1];
  const float* conv_W[3] = { (const float*)d_in[2], (const float*)d_in[4], (const float*)d_in[6] };
  const float* conv_b[3] = { (const float*)d_in[3], (const float*)d_in[5], (const float*)d_in[7] };
  const float* lin1_W = (const float*)d_in[8];
  const float* lin1_b = (const float*)d_in[9];
  const float* lin2_W = (const float*)d_in[10];
  const float* lin2_b = (const float*)d_in[11];
  const float* gn_w[3] = { (const float*)d_in[12], (const float*)d_in[15], (const float*)d_in[18] };
  const float* gn_b[3] = { (const float*)d_in[13], (const float*)d_in[16], (const float*)d_in[19] };
  const float* gn_a[3] = { (const float*)d_in[14], (const float*)d_in[17], (const float*)d_in[20] };
  const int* srcI = ei;
  const int* dstI = ei + EE;

  auto cdiv = [](long a, long b) { return (int)((a + b - 1) / b); };

  // padded dims
  const int Kp0    = cdiv(INF_, BK) * BK;   // 4672  (conv0 K)
  const int KpH    = HF;                    // 512 already multiple of 32
  const int NcPadH = cdiv(HF, BN) * BN;     // 512
  const int NcPadO = cdiv(INF_, BN) * BN;   // 4736  (lin2 Nc)

  // ---- workspace carve ----
  char* p = (char*)d_ws;
  auto carve = [&](size_t bytes) -> void* {
    void* r = (void*)p;
    p += (bytes + 255) & ~(size_t)255;
    return r;
  };
  __bf16* w_bfT[3];
  w_bfT[0]        = (__bf16*)carve((size_t)NcPadH * Kp0 * 2);
  w_bfT[1]        = (__bf16*)carve((size_t)NcPadH * KpH * 2);
  w_bfT[2]        = (__bf16*)carve((size_t)NcPadH * KpH * 2);
  __bf16* l1w_bfT = (__bf16*)carve((size_t)NcPadH * KpH * 2);
  __bf16* l2w_bfT = (__bf16*)carve((size_t)NcPadO * KpH * 2);
  float*  hW      = (float*)carve((size_t)NN * HF * 4);
  float*  agg     = (float*)carve((size_t)NN * HF * 4);   // also y (in place)
  float*  lay[3];
  lay[0]          = (float*)carve((size_t)NN * HF * 4);
  lay[1]          = (float*)carve((size_t)NN * HF * 4);
  lay[2]          = (float*)carve((size_t)NN * HF * 4);
  __bf16* hbf     = (__bf16*)carve((size_t)NN * HF * 2);  // conv input / jk
  __bf16* tbf     = (__bf16*)carve((size_t)NN * HF * 2);  // lin1 output
  float*  deg     = (float*)carve((size_t)NN * 4);
  float*  dinv    = (float*)carve((size_t)NN * 4);
  float*  dinv2   = (float*)carve((size_t)NN * 4);
  float*  coef    = (float*)carve((size_t)EE * 4);
  float*  colsum  = (float*)carve((size_t)HF * 4);
  float*  meanB   = (float*)carve((size_t)HF * 4);
  float*  varsum  = (float*)carve((size_t)HF * 4);
  float*  rstdB   = (float*)carve((size_t)HF * 4);

  // ---- weight convert + transpose + pad ----
  k_f2bf_T<<<cdiv((long)NcPadH * Kp0, 256), 256, 0, stream>>>(conv_W[0], w_bfT[0], INF_, HF, Kp0, NcPadH);
  k_f2bf_T<<<cdiv((long)NcPadH * KpH, 256), 256, 0, stream>>>(conv_W[1], w_bfT[1], HF, HF, KpH, NcPadH);
  k_f2bf_T<<<cdiv((long)NcPadH * KpH, 256), 256, 0, stream>>>(conv_W[2], w_bfT[2], HF, HF, KpH, NcPadH);
  k_f2bf_T<<<cdiv((long)NcPadH * KpH, 256), 256, 0, stream>>>(lin1_W,    l1w_bfT, HF, HF, KpH, NcPadH);
  k_f2bf_T<<<cdiv((long)NcPadO * KpH, 256), 256, 0, stream>>>(lin2_W,    l2w_bfT, HF, INF_, KpH, NcPadO);

  // ---- degree / normalization coefficients ----
  hipMemsetAsync(deg, 0, (size_t)NN * 4, stream);
  k_degree<<<cdiv(EE, 256), 256, 0, stream>>>(dstI, deg, EE);
  k_dinv  <<<cdiv(NN, 256), 256, 0, stream>>>(deg, dinv, dinv2, NN);
  k_coef  <<<cdiv(EE, 256), 256, 0, stream>>>(srcI, dstI, dinv, coef, EE);

  const long NH = (long)NN * HF;
  const dim3 gemmBlk(256);
  const dim3 gH(NcPadH / BN, cdiv(NN, BM));   // (4, 157)
  const dim3 gO(NcPadO / BN, cdiv(NN, BM));   // (37, 157)

  for (int i = 0; i < 3; ++i) {
    if (i == 0)
      gemm_wmma_bf16<float><<<gH, gemmBlk, 0, stream>>>(
          x, w_bfT[0], NN, INF_, Kp0, HF, nullptr, 0, hW, nullptr);
    else
      gemm_wmma_bf16<__bf16><<<gH, gemmBlk, 0, stream>>>(
          hbf, w_bfT[i], NN, HF, KpH, HF, nullptr, 0, hW, nullptr);

    hipMemsetAsync(agg, 0, (size_t)NH * 4, stream);
    k_scatter<<<EE, 256, 0, stream>>>(srcI, dstI, coef, hW, agg, EE);

    hipMemsetAsync(colsum, 0, (size_t)HF * 4, stream);
    k_combine_colsum<<<cdiv(NN, 32), 256, 0, stream>>>(agg, hW, dinv2, conv_b[i], colsum, NN);
    k_mean<<<cdiv(HF, 256), 256, 0, stream>>>(colsum, meanB);
    hipMemsetAsync(varsum, 0, (size_t)HF * 4, stream);
    k_varsum<<<cdiv(NN, 32), 256, 0, stream>>>(agg, meanB, gn_a[i], varsum, NN);
    k_rstd<<<cdiv(HF, 256), 256, 0, stream>>>(varsum, rstdB);

    const float* res = (i == 1) ? lay[0] : nullptr;   // layer-2 next-h is dead
    __bf16* nxt = (i < 2) ? hbf : nullptr;
    k_norm_final<<<2048, 256, 0, stream>>>(agg, meanB, rstdB, gn_a[i], gn_w[i], gn_b[i],
                                           res, lay[i], nxt, NH);
  }

  // JumpingKnowledge max -> bf16
  k_jkmax_bf<<<2048, 256, 0, stream>>>(lay[0], lay[1], lay[2], hbf, NH);

  // lin1: relu(jk @ W + b) -> bf16
  gemm_wmma_bf16<__bf16><<<gH, gemmBlk, 0, stream>>>(
      hbf, l1w_bfT, NN, HF, KpH, HF, lin1_b, 1, nullptr, tbf);

  // lin2: t @ W + b -> f32 output
  gemm_wmma_bf16<__bf16><<<gO, gemmBlk, 0, stream>>>(
      tbf, l2w_bfT, NN, HF, KpH, INF_, lin2_b, 0, (float*)d_out, nullptr);
}
